// GraphModule_30820685316765
// MI455X (gfx1250) — compile-verified, matrix operand order checked by reading, not run on hardware
//
#include <hip/hip_runtime.h>
#include <math.h>

// ---------------- model dims ----------------
#define B_   16
#define S_   512
#define D_   768
#define H_   12
#define FF_  3072
#define L_   4
#define DH_  64     // D/H

typedef __bf16 bhalf;
typedef __attribute__((ext_vector_type(16))) __bf16 v16bf;
typedef __attribute__((ext_vector_type(8)))  __bf16 v8bf;
typedef __attribute__((ext_vector_type(8)))  float  v8f;

// ---------------- reductions ----------------
__device__ __forceinline__ float block_reduce_sum(float v, float* red) {
  int t = threadIdx.x;
  red[t] = v; __syncthreads();
  for (int s = blockDim.x >> 1; s > 0; s >>= 1) {
    if (t < s) red[t] += red[t + s];
    __syncthreads();
  }
  float r = red[0]; __syncthreads();
  return r;
}

__device__ __forceinline__ float block_reduce_max(float v, float* red) {
  int t = threadIdx.x;
  red[t] = v; __syncthreads();
  for (int s = blockDim.x >> 1; s > 0; s >>= 1) {
    if (t < s) red[t] = fmaxf(red[t], red[t + s]);
    __syncthreads();
  }
  float r = red[0]; __syncthreads();
  return r;
}

// ---------------- f32 -> bf16 convert ----------------
__global__ void cvt_bf16_kernel(const float* __restrict__ in, bhalf* __restrict__ out, long n) {
  long i = (long)blockIdx.x * blockDim.x + threadIdx.x;
  if (i < n) out[i] = (bhalf)in[i];
}

// ---------------- embedding + LayerNorm ----------------
__global__ void embed_ln_kernel(const int* __restrict__ ids, const int* __restrict__ tts,
                                const float* __restrict__ wemb, const float* __restrict__ temb,
                                const float* __restrict__ lw, const float* __restrict__ lb,
                                float* __restrict__ out32, bhalf* __restrict__ outb) {
  __shared__ float red[256];
  long row = blockIdx.x;                      // b*S + s
  const float* wp = wemb + (long)ids[row] * D_;
  const float* tp = temb + (long)tts[row] * D_;
  float x[3], lsum = 0.f;
  #pragma unroll
  for (int i = 0; i < 3; ++i) {
    int d = threadIdx.x + i * 256;
    x[i] = wp[d] + tp[d];
    lsum += x[i];
  }
  float mean = block_reduce_sum(lsum, red) * (1.0f / D_);
  float lvar = 0.f;
  #pragma unroll
  for (int i = 0; i < 3; ++i) { float c = x[i] - mean; lvar += c * c; }
  float inv = rsqrtf(block_reduce_sum(lvar, red) * (1.0f / D_) + 1e-12f);
  #pragma unroll
  for (int i = 0; i < 3; ++i) {
    int d = threadIdx.x + i * 256;
    float y = (x[i] - mean) * inv * lw[d] + lb[d];
    out32[row * D_ + d] = y;
    outb[row * D_ + d]  = (bhalf)y;
  }
}

// ---------------- residual add + LayerNorm ----------------
__global__ void add_ln_kernel(const float* __restrict__ a, const float* __restrict__ res,
                              const float* __restrict__ lw, const float* __restrict__ lb,
                              float* __restrict__ out32, bhalf* __restrict__ outb) {
  __shared__ float red[256];
  long row = blockIdx.x;
  float x[3], lsum = 0.f;
  #pragma unroll
  for (int i = 0; i < 3; ++i) {
    int d = threadIdx.x + i * 256;
    x[i] = a[row * D_ + d] + res[row * D_ + d];
    lsum += x[i];
  }
  float mean = block_reduce_sum(lsum, red) * (1.0f / D_);
  float lvar = 0.f;
  #pragma unroll
  for (int i = 0; i < 3; ++i) { float c = x[i] - mean; lvar += c * c; }
  float inv = rsqrtf(block_reduce_sum(lvar, red) * (1.0f / D_) + 1e-12f);
  #pragma unroll
  for (int i = 0; i < 3; ++i) {
    int d = threadIdx.x + i * 256;
    float y = (x[i] - mean) * inv * lw[d] + lb[d];
    out32[row * D_ + d] = y;
    outb[row * D_ + d]  = (bhalf)y;
  }
}

// ---------------- exact GELU (f32 in -> bf16 out) ----------------
__global__ void gelu_kernel(const float* __restrict__ in, bhalf* __restrict__ out, long n) {
  long i = (long)blockIdx.x * blockDim.x + threadIdx.x;
  if (i < n) {
    float x = in[i];
    out[i] = (bhalf)(0.5f * x * (1.0f + erff(x * 0.70710678118654752f)));
  }
}

// ---------------- softmax over scores (scale + mask bias), writes bf16 probs ----------------
__global__ void softmax_kernel(const float* __restrict__ sc, const int* __restrict__ mask,
                               bhalf* __restrict__ probs) {
  __shared__ float red[256];
  long row = blockIdx.x;                       // b*H*S + h*S + q
  int b = (int)(row / ((long)H_ * S_));
  const float* sp = sc + row * S_;
  const int* mp = mask + (long)b * S_;
  float v[2], mx = -3.0e38f;
  #pragma unroll
  for (int i = 0; i < 2; ++i) {
    int k = threadIdx.x + i * 256;
    float bias = (1.0f - (float)mp[k]) * -10000.0f;
    v[i] = sp[k] * 0.125f + bias;              // 1/sqrt(DH) = 0.125
    mx = fmaxf(mx, v[i]);
  }
  mx = block_reduce_max(mx, red);
  float ls = 0.f;
  #pragma unroll
  for (int i = 0; i < 2; ++i) { v[i] = __expf(v[i] - mx); ls += v[i]; }
  float inv = 1.0f / block_reduce_sum(ls, red);
  #pragma unroll
  for (int i = 0; i < 2; ++i) {
    int k = threadIdx.x + i * 256;
    probs[row * S_ + k] = (bhalf)(v[i] * inv);
  }
}

// ---------------- head split / transpose kernels ----------------
__global__ void split_heads_kernel(const float* __restrict__ in, bhalf* __restrict__ out) {
  long o = (long)blockIdx.x * blockDim.x + threadIdx.x;      // (b,h,s,d)
  const long total = (long)B_ * H_ * S_ * DH_;
  if (o >= total) return;
  int d = (int)(o % DH_);
  long t = o / DH_;
  int s = (int)(t % S_); t /= S_;
  int h = (int)(t % H_);
  int b = (int)(t / H_);
  out[o] = (bhalf)in[((long)b * S_ + s) * D_ + h * DH_ + d];
}

// v (B,S,D) f32  ->  vhT (B,H,DH,S) bf16   (k-contiguous "W" operand for P@V)
__global__ void transpose_v_kernel(const float* __restrict__ in, bhalf* __restrict__ out) {
  long o = (long)blockIdx.x * blockDim.x + threadIdx.x;
  const long total = (long)B_ * H_ * DH_ * S_;
  if (o >= total) return;
  int k = (int)(o % S_);
  long t = o / S_;
  int d = (int)(t % DH_); t /= DH_;
  int h = (int)(t % H_);
  int b = (int)(t / H_);
  out[o] = (bhalf)in[((long)b * S_ + k) * D_ + h * DH_ + d];
}

// rel (S,S,DH) f32 -> relT (S,DH,S) bf16   (k-contiguous "W" operand for P@rel)
__global__ void transpose_rel_kernel(const float* __restrict__ in, bhalf* __restrict__ out) {
  long o = (long)blockIdx.x * blockDim.x + threadIdx.x;
  const long total = (long)S_ * DH_ * S_;
  if (o >= total) return;
  int k = (int)(o % S_);
  long t = o / S_;
  int d = (int)(t % DH_);
  int q = (int)(t / DH_);
  out[o] = (bhalf)in[((long)q * S_ + k) * DH_ + d];
}

// merge: ctx (B,H,S,DH) f32 -> (B,S,D) bf16
__global__ void merge_heads_kernel(const float* __restrict__ in, bhalf* __restrict__ out) {
  long o = (long)blockIdx.x * blockDim.x + threadIdx.x;
  const long total = (long)B_ * S_ * D_;
  if (o >= total) return;
  int hd = (int)(o % D_);
  int h = hd / DH_, d = hd % DH_;
  long t = o / D_;
  int s = (int)(t % S_);
  int b = (int)(t / S_);
  out[o] = (bhalf)in[(((long)b * H_ + h) * S_ + s) * DH_ + d];
}

// ---------------- fragment load (ISA 16-bit layout; 2x16B vector loads) ----------------
__device__ __forceinline__ v16bf load_frag(const bhalf* p) {
  v8bf lo = *(const v8bf*)p;         // K kk..kk+7
  v8bf hi = *(const v8bf*)(p + 16);  // K kk+16..kk+23
  return __builtin_shufflevector(lo, hi, 0, 1, 2, 3, 4, 5, 6, 7,
                                         8, 9, 10, 11, 12, 13, 14, 15);
}

// ---------------- batched WMMA GEMM:  C[z] (+)= A[z] * W[z]^T (+bias) ----------------
// Register-blocked 32x64 per wave: 2 A-frags x 4 W-frags -> 8 wmma per K-32 step;
// every W fragment is reused twice, every A fragment four times (~21 FLOP/B from L2).
// A: bf16, element (m,k) at A + aStride*z + m*lda + k     (k contiguous, 16B aligned rows)
// W: bf16, element (n,k) at W + wStride*z + n*wns + k     (k contiguous, 16B aligned rows)
// C: f32,  element (m,n) at C + cStride*z + m*ldc + n
__global__ void wmma_gemm_kernel(const bhalf* __restrict__ A, const bhalf* __restrict__ W,
                                 const float* __restrict__ bias, float* __restrict__ C,
                                 int M, int N, int K,
                                 long lda, long wns, long ldc,
                                 long aStride, long wStride, long cStride,
                                 int accumulate) {
  int lane = threadIdx.x & 31;
  int wave = threadIdx.x >> 5;
  int groupsN = N >> 6;                         // 64-wide N groups
  int tiles   = (M >> 5) * groupsN;             // 32-row M tiles
  int tile = blockIdx.x * (blockDim.x >> 5) + wave;
  if (tile >= tiles) return;                    // whole-wave uniform exit; EXEC all-1 for WMMA
  int m0 = (tile / groupsN) << 5;
  int n0 = (tile % groupsN) << 6;
  long z = blockIdx.y;

  int half = lane >> 4;                         // K-phase per ISA 16-bit A/B layout
  int r    = lane & 15;                         // row (A) / column (B,C)

  const bhalf* ap0 = A + aStride * z + (long)(m0 + r) * lda + half * 8;
  const bhalf* ap1 = ap0 + 16 * lda;
  const bhalf* wp0 = W + wStride * z + (long)(n0 + r) * wns + half * 8;
  const bhalf* wp1 = wp0 + 16 * wns;
  const bhalf* wp2 = wp0 + 32 * wns;
  const bhalf* wp3 = wp0 + 48 * wns;
  float* Cz = C + cStride * z;

  v8f acc00, acc01, acc02, acc03, acc10, acc11, acc12, acc13;
  #pragma unroll
  for (int v = 0; v < 8; ++v) {
    acc00[v] = 0.f; acc01[v] = 0.f; acc02[v] = 0.f; acc03[v] = 0.f;
    acc10[v] = 0.f; acc11[v] = 0.f; acc12[v] = 0.f; acc13[v] = 0.f;
  }
  if (accumulate) {
    #pragma unroll
    for (int v = 0; v < 8; ++v) {
      long ro0 = (long)(m0 + half * 8 + v) * ldc + r;
      long ro1 = ro0 + 16 * ldc;
      acc00[v] = Cz[ro0 + n0];      acc01[v] = Cz[ro0 + n0 + 16];
      acc02[v] = Cz[ro0 + n0 + 32]; acc03[v] = Cz[ro0 + n0 + 48];
      acc10[v] = Cz[ro1 + n0];      acc11[v] = Cz[ro1 + n0 + 16];
      acc12[v] = Cz[ro1 + n0 + 32]; acc13[v] = Cz[ro1 + n0 + 48];
    }
  }

  #pragma unroll 2
  for (int k0 = 0; k0 < K; k0 += 32) {
    v16bf af0 = load_frag(ap0);
    v16bf af1 = load_frag(ap1);
    v16bf wf0 = load_frag(wp0);
    v16bf wf1 = load_frag(wp1);
    v16bf wf2 = load_frag(wp2);
    v16bf wf3 = load_frag(wp3);
    __builtin_prefetch(ap0 + 64, 0, 1);         // CDNA5 global_prefetch_b8
    __builtin_prefetch(wp0 + 64, 0, 1);

    acc00 = __builtin_amdgcn_wmma_f32_16x16x32_bf16(false, af0, false, wf0, (short)0, acc00, false, false);
    acc01 = __builtin_amdgcn_wmma_f32_16x16x32_bf16(false, af0, false, wf1, (short)0, acc01, false, false);
    acc02 = __builtin_amdgcn_wmma_f32_16x16x32_bf16(false, af0, false, wf2, (short)0, acc02, false, false);
    acc03 = __builtin_amdgcn_wmma_f32_16x16x32_bf16(false, af0, false, wf3, (short)0, acc03, false, false);
    acc10 = __builtin_amdgcn_wmma_f32_16x16x32_bf16(false, af1, false, wf0, (short)0, acc10, false, false);
    acc11 = __builtin_amdgcn_wmma_f32_16x16x32_bf16(false, af1, false, wf1, (short)0, acc11, false, false);
    acc12 = __builtin_amdgcn_wmma_f32_16x16x32_bf16(false, af1, false, wf2, (short)0, acc12, false, false);
    acc13 = __builtin_amdgcn_wmma_f32_16x16x32_bf16(false, af1, false, wf3, (short)0, acc13, false, false);

    ap0 += 32; ap1 += 32; wp0 += 32; wp1 += 32; wp2 += 32; wp3 += 32;
  }

  float b0 = 0.f, b1 = 0.f, b2 = 0.f, b3 = 0.f;
  if (bias) {
    b0 = bias[n0 + r]; b1 = bias[n0 + 16 + r]; b2 = bias[n0 + 32 + r]; b3 = bias[n0 + 48 + r];
  }
  #pragma unroll
  for (int v = 0; v < 8; ++v) {
    long ro0 = (long)(m0 + half * 8 + v) * ldc + r;
    long ro1 = ro0 + 16 * ldc;
    Cz[ro0 + n0]      = acc00[v] + b0;
    Cz[ro0 + n0 + 16] = acc01[v] + b1;
    Cz[ro0 + n0 + 32] = acc02[v] + b2;
    Cz[ro0 + n0 + 48] = acc03[v] + b3;
    Cz[ro1 + n0]      = acc10[v] + b0;
    Cz[ro1 + n0 + 16] = acc11[v] + b1;
    Cz[ro1 + n0 + 32] = acc12[v] + b2;
    Cz[ro1 + n0 + 48] = acc13[v] + b3;
  }
}

// ---------------- host-side helpers ----------------
static inline void launch_gemm(hipStream_t s, const bhalf* A, const bhalf* W, const float* bias,
                               float* C, int M, int N, int K,
                               long lda, long wns, long ldc,
                               int batches, long aStride, long wStride, long cStride,
                               int accumulate) {
  int tiles = (M / 32) * (N / 64);
  dim3 grid((tiles + 3) / 4, batches);
  wmma_gemm_kernel<<<grid, 128, 0, s>>>(A, W, bias, C, M, N, K, lda, wns, ldc,
                                        aStride, wStride, cStride, accumulate);
}

static inline void launch_cvt(hipStream_t s, const float* in, bhalf* out, long n) {
  cvt_bf16_kernel<<<(unsigned)((n + 255) / 256), 256, 0, s>>>(in, out, n);
}

extern "C" void kernel_launch(void* const* d_in, const int* in_sizes, int n_in,
                              void* d_out, int out_size, void* d_ws, size_t ws_size,
                              hipStream_t stream) {
  (void)in_sizes; (void)n_in; (void)out_size; (void)ws_size;

  // ---- inputs ----
  const int*   ids   = (const int*)d_in[0];
  const int*   amask = (const int*)d_in[1];
  const int*   tts   = (const int*)d_in[2];
  const float* wemb  = (const float*)d_in[3];
  const float* temb  = (const float*)d_in[4];
  const float* elnw  = (const float*)d_in[5];
  const float* elnb  = (const float*)d_in[6];
  const float* qw    = (const float*)d_in[7];
  const float* qb    = (const float*)d_in[8];
  const float* kw    = (const float*)d_in[9];
  const float* kb    = (const float*)d_in[10];
  const float* vw    = (const float*)d_in[11];
  const float* vb    = (const float*)d_in[12];
  const float* rel   = (const float*)d_in[13];
  const float* ow    = (const float*)d_in[14];
  const float* ob    = (const float*)d_in[15];
  const float* alnw  = (const float*)d_in[16];
  const float* alnb  = (const float*)d_in[17];
  const float* iw    = (const float*)d_in[18];
  const float* ib    = (const float*)d_in[19];
  const float* dw    = (const float*)d_in[20];
  const float* db    = (const float*)d_in[21];
  const float* olnw  = (const float*)d_in[22];
  const float* olnb  = (const float*)d_in[23];

  // ---- element counts ----
  const long ND   = (long)B_ * S_ * D_;        // 6.29M
  const long NS   = (long)B_ * H_ * S_ * S_;   // 50.3M
  const long NFF  = (long)B_ * S_ * FF_;       // 25.2M
  const long WQ   = (long)L_ * D_ * D_;
  const long WI   = (long)L_ * FF_ * D_;
  const long NREL = (long)S_ * S_ * DH_;

  // ---- workspace carve-out ----
  char* p = (char*)d_ws;
  auto alloc = [&](long bytes) { char* r = p; p += (bytes + 255) & ~255L; return r; };
  bhalf* qwb  = (bhalf*)alloc(WQ * 2);
  bhalf* kwb  = (bhalf*)alloc(WQ * 2);
  bhalf* vwb  = (bhalf*)alloc(WQ * 2);
  bhalf* owb  = (bhalf*)alloc(WQ * 2);
  bhalf* iwb  = (bhalf*)alloc(WI * 2);
  bhalf* dwb  = (bhalf*)alloc(WI * 2);
  bhalf* relb = (bhalf*)alloc(NREL * 2);       // (q,k,d) for q@rel^T scores
  bhalf* relT = (bhalf*)alloc(NREL * 2);       // (q,d,k) for P@rel context
  float* h32  = (float*)alloc(ND * 4);
  bhalf* hbf  = (bhalf*)alloc(ND * 2);
  float* q32  = (float*)alloc(ND * 4);
  float* k32  = (float*)alloc(ND * 4);
  float* v32  = (float*)alloc(ND * 4);
  bhalf* qhb  = (bhalf*)alloc(ND * 2);
  bhalf* khb  = (bhalf*)alloc(ND * 2);
  bhalf* vhT  = (bhalf*)alloc(ND * 2);         // (b,h,d,k)
  float* sc32 = (float*)alloc(NS * 4);
  bhalf* pb   = (bhalf*)alloc(NS * 2);
  float* ctx32= (float*)alloc(ND * 4);
  bhalf* cmb  = (bhalf*)alloc(ND * 2);
  float* proj = (float*)alloc(ND * 4);
  float* ao32 = (float*)alloc(ND * 4);
  bhalf* aob  = (bhalf*)alloc(ND * 2);
  float* ff32 = (float*)alloc(NFF * 4);
  bhalf* ffb  = (bhalf*)alloc(NFF * 2);
  float* ffo  = (float*)alloc(ND * 4);

  // ---- weight conversion / rel transposes (every call; deterministic) ----
  launch_cvt(stream, qw, qwb, WQ);
  launch_cvt(stream, kw, kwb, WQ);
  launch_cvt(stream, vw, vwb, WQ);
  launch_cvt(stream, ow, owb, WQ);
  launch_cvt(stream, iw, iwb, WI);
  launch_cvt(stream, dw, dwb, WI);
  launch_cvt(stream, rel, relb, NREL);
  transpose_rel_kernel<<<(unsigned)((NREL + 255) / 256), 256, 0, stream>>>(rel, relT);

  // ---- embeddings + LN ----
  embed_ln_kernel<<<B_ * S_, 256, 0, stream>>>(ids, tts, wemb, temb, elnw, elnb, h32, hbf);

  const int  Mrow = B_ * S_;                    // 8192
  const int  BH   = B_ * H_;                    // 192
  const long SDH  = (long)S_ * DH_;
  const long SS   = (long)S_ * S_;
  const long DHS  = (long)DH_ * S_;

  for (int l = 0; l < L_; ++l) {
    const long wq = (long)l * D_ * D_;
    const long wi = (long)l * FF_ * D_;
    const long ld = (long)l * D_;

    // Q/K/V projections  (x @ W^T + b)
    launch_gemm(stream, hbf, qwb + wq, qb + ld, q32, Mrow, D_, D_, D_, D_, D_, 1, 0, 0, 0, 0);
    launch_gemm(stream, hbf, kwb + wq, kb + ld, k32, Mrow, D_, D_, D_, D_, D_, 1, 0, 0, 0, 0);
    launch_gemm(stream, hbf, vwb + wq, vb + ld, v32, Mrow, D_, D_, D_, D_, D_, 1, 0, 0, 0, 0);

    unsigned gsh = (unsigned)((ND + 255) / 256);
    split_heads_kernel<<<gsh, 256, 0, stream>>>(q32, qhb);
    split_heads_kernel<<<gsh, 256, 0, stream>>>(k32, khb);
    transpose_v_kernel<<<gsh, 256, 0, stream>>>(v32, vhT);

    // content scores: per (b,h): q(SxDH) @ k^T  -> sc[bh, q, ki]
    launch_gemm(stream, qhb, khb, nullptr, sc32, S_, S_, DH_,
                DH_, DH_, S_, BH, SDH, SDH, SS, 0);
    // rel scores (accumulate): per q: q_bh(BHxDH) @ rel[q]^T -> sc[bh, q, ki]
    launch_gemm(stream, qhb, relb, nullptr, sc32, BH, S_, DH_,
                SDH, DH_, SS, 512, DH_, SDH, S_, 1);

    // softmax (scale + mask bias) -> bf16 probs
    softmax_kernel<<<(unsigned)((long)BH * S_), 256, 0, stream>>>(sc32, amask, pb);

    // content context: per (b,h): probs(SxS) @ v -> ctx[bh, q, d]   (W = vhT, k-contiguous)
    launch_gemm(stream, pb, vhT, nullptr, ctx32, S_, DH_, S_,
                S_, S_, DH_, BH, SS, DHS, SDH, 0);
    // rel context (accumulate): per q: probs_bh(BHxS) @ rel[q] -> ctx[bh, q, d]  (W = relT)
    launch_gemm(stream, pb, relT, nullptr, ctx32, BH, DH_, S_,
                SS, S_, SDH, 512, S_, DHS, DH_, 1);

    merge_heads_kernel<<<gsh, 256, 0, stream>>>(ctx32, cmb);

    // attention output projection + residual + LN
    launch_gemm(stream, cmb, owb + wq, ob + ld, proj, Mrow, D_, D_, D_, D_, D_, 1, 0, 0, 0, 0);
    add_ln_kernel<<<Mrow, 256, 0, stream>>>(proj, h32, alnw + ld, alnb + ld, ao32, aob);

    // FFN
    launch_gemm(stream, aob, iwb + wi, ib + (long)l * FF_, ff32,
                Mrow, FF_, D_, D_, D_, FF_, 1, 0, 0, 0, 0);
    gelu_kernel<<<(unsigned)((NFF + 255) / 256), 256, 0, stream>>>(ff32, ffb, NFF);
    launch_gemm(stream, ffb, dwb + wi, db + ld, ffo,
                Mrow, D_, FF_, FF_, FF_, D_, 1, 0, 0, 0, 0);

    float* hnext = (l == L_ - 1) ? (float*)d_out : h32;
    add_ln_kernel<<<Mrow, 256, 0, stream>>>(ffo, ao32, olnw + ld, olnb + ld, hnext, hbf);
  }
}
